// CARAFE_5420248727930
// MI455X (gfx1250) — matrix operand order repaired and checked by reading, not compiled, not run on hardware
//
#include <hip/hip_runtime.h>
#include <hip/hip_bf16.h>

typedef __attribute__((ext_vector_type(16))) __bf16 v16bf;
typedef __attribute__((ext_vector_type(8)))  float  v8f;

__device__ __forceinline__ __bf16 f2bf(float f) {
    union { float f; unsigned u; } v; v.f = f;
    union { unsigned short s; __bf16 b; } o;
    o.s = (unsigned short)(v.u >> 16);
    return o.b;
}

// ---------------------------------------------------------------------------
// Kernel 1: 1x1 conv (channel compress 256 -> 64) as bf16 WMMA GEMM.
//   Per batch: Wc[64,16384] = comp_w[64,256] * X[256,16384]
//   One wave computes a 16x16 output tile; K stepped in chunks of 32.
// ---------------------------------------------------------------------------
__global__ __launch_bounds__(256) void carafe_conv1x1_wmma(
    const float* __restrict__ X, const float* __restrict__ W,
    const float* __restrict__ bias, float* __restrict__ Wc)
{
    const int wid  = (blockIdx.x * blockDim.x + threadIdx.x) >> 5;
    const int lane = threadIdx.x & 31;
    const int lm   = lane & 15;
    const int hi   = lane >> 4;

    const int nt   = wid & 1023;       // 16384/16 N-tiles
    const int rest = wid >> 10;
    const int mt   = rest & 3;         // 64/16 M-tiles
    const int b    = rest >> 2;        // batch

    const int m0 = mt << 4;
    const int n0 = nt << 4;
    const int n  = n0 + lm;

    v8f acc = {};
    const float* arow = W + (size_t)(m0 + lm) * 256;
    const float* xb   = X + (size_t)b * 256 * 16384 + n;

    #pragma unroll 2
    for (int kc = 0; kc < 8; ++kc) {
        const int kbase = kc << 5;
        // A (16x32 bf16): lane row = m0+lm; elem e -> K = kbase + hi*16 + e
        // (same contiguous K order as B).
        const float* ap = arow + kbase + (hi << 4);
        float4 a0 = *(const float4*)(ap);
        float4 a1 = *(const float4*)(ap + 4);
        float4 a2 = *(const float4*)(ap + 8);
        float4 a3 = *(const float4*)(ap + 12);
        v16bf A;
        A[0]=f2bf(a0.x);  A[1]=f2bf(a0.y);  A[2]=f2bf(a0.z);  A[3]=f2bf(a0.w);
        A[4]=f2bf(a1.x);  A[5]=f2bf(a1.y);  A[6]=f2bf(a1.z);  A[7]=f2bf(a1.w);
        A[8]=f2bf(a2.x);  A[9]=f2bf(a2.y);  A[10]=f2bf(a2.z); A[11]=f2bf(a2.w);
        A[12]=f2bf(a3.x); A[13]=f2bf(a3.y); A[14]=f2bf(a3.z); A[15]=f2bf(a3.w);
        // B (32x16 bf16): lane col = n0+lm; elem e -> K = kbase + hi*16 + e.
        v16bf B;
        const int kb = kbase + (hi << 4);
        #pragma unroll
        for (int e = 0; e < 16; ++e)
            B[e] = f2bf(xb[(size_t)(kb + e) * 16384]);
        acc = __builtin_amdgcn_wmma_f32_16x16x32_bf16(false, A, false, B,
                                                      (short)0, acc, false, false);
    }

    const int mbase = m0 + (hi << 3);
    float* outp = Wc + ((size_t)b * 64 + mbase) * 16384 + n;
    #pragma unroll
    for (int r = 0; r < 8; ++r)
        outp[(size_t)r * 16384] = acc[r] + bias[mbase + r];
}

// ---------------------------------------------------------------------------
// Kernel 2: 3x3 conv (64 -> 100 ch, pad 1) as bf16 WMMA GEMM with on-the-fly
// im2col. M padded to 112 (7 tiles); K = 64*9 = 576 = 18 chunks of 32.
// ---------------------------------------------------------------------------
__global__ __launch_bounds__(256) void carafe_conv3x3_wmma(
    const float* __restrict__ Wc, const float* __restrict__ W,
    const float* __restrict__ bias, float* __restrict__ We)
{
    const int wid  = (blockIdx.x * blockDim.x + threadIdx.x) >> 5;
    const int lane = threadIdx.x & 31;
    const int lm   = lane & 15;
    const int hi   = lane >> 4;

    const int nt   = wid & 1023;
    const int rest = wid >> 10;
    const int mt   = rest % 7;
    const int b    = rest / 7;

    const int m0 = mt << 4;
    const int n0 = nt << 4;
    const int n  = n0 + lm;
    const int y  = n >> 7;
    const int x  = n & 127;

    const bool arow_ok = (m0 + lm) < 100;
    const float* arow = W + (size_t)(m0 + lm) * 576;
    const float* cb   = Wc + (size_t)b * 64 * 16384;

    v8f acc = {};
    for (int kc = 0; kc < 18; ++kc) {
        const int kbase = kc << 5;
        v16bf A;
        if (arow_ok) {
            const float* ap = arow + kbase + (hi << 4);
            float4 a0 = *(const float4*)(ap);
            float4 a1 = *(const float4*)(ap + 4);
            float4 a2 = *(const float4*)(ap + 8);
            float4 a3 = *(const float4*)(ap + 12);
            A[0]=f2bf(a0.x);  A[1]=f2bf(a0.y);  A[2]=f2bf(a0.z);  A[3]=f2bf(a0.w);
            A[4]=f2bf(a1.x);  A[5]=f2bf(a1.y);  A[6]=f2bf(a1.z);  A[7]=f2bf(a1.w);
            A[8]=f2bf(a2.x);  A[9]=f2bf(a2.y);  A[10]=f2bf(a2.z); A[11]=f2bf(a2.w);
            A[12]=f2bf(a3.x); A[13]=f2bf(a3.y); A[14]=f2bf(a3.z); A[15]=f2bf(a3.w);
        } else {
            #pragma unroll
            for (int e = 0; e < 16; ++e) A[e] = f2bf(0.0f);
        }
        // B: im2col gather. k -> (c = k/9, tap = k%9 -> dy,dx in {-1,0,1}).
        v16bf B;
        const int kb = kbase + (hi << 4);
        #pragma unroll
        for (int e = 0; e < 16; ++e) {
            const int k  = kb + e;
            const int c  = k / 9;
            const int r9 = k - c * 9;
            const int dy = r9 / 3 - 1;
            const int dx = r9 % 3 - 1;
            const int yy = y + dy, xx = x + dx;
            float v = 0.0f;
            if ((unsigned)yy < 128u && (unsigned)xx < 128u)
                v = cb[(size_t)c * 16384 + yy * 128 + xx];
            B[e] = f2bf(v);
        }
        acc = __builtin_amdgcn_wmma_f32_16x16x32_bf16(false, A, false, B,
                                                      (short)0, acc, false, false);
    }

    const int mbase = m0 + (hi << 3);
    #pragma unroll
    for (int r = 0; r < 8; ++r) {
        const int m = mbase + r;
        if (m < 100)
            We[((size_t)b * 100 + m) * 16384 + n] = acc[r] + bias[m];
    }
}

// ---------------------------------------------------------------------------
// Kernel 3: pixel-shuffle + softmax over 25 taps.
// We (b,100,128,128), channel = kk*4 + si*2 + sj -> Wn stored as (b,256,256,25)
// so the reassembly kernel reads its 25 weights contiguously.
// ---------------------------------------------------------------------------
__global__ __launch_bounds__(256) void carafe_softmax(
    const float* __restrict__ We, float* __restrict__ Wn)
{
    const int tid = blockIdx.x * blockDim.x + threadIdx.x;   // 2^18 threads
    const int n   = tid & 16383;
    const int off = (tid >> 14) & 3;
    const int b   = tid >> 16;
    const int y = n >> 7, x = n & 127;
    const int si = off >> 1, sj = off & 1;

    const float* src = We + ((size_t)b * 100 + off) * 16384 + n;
    float v[25];
    float mx = -3.4e38f;
    #pragma unroll
    for (int kk = 0; kk < 25; ++kk) {
        v[kk] = src[(size_t)kk * 4 * 16384];
        mx = fmaxf(mx, v[kk]);
    }
    float s = 0.0f;
    #pragma unroll
    for (int kk = 0; kk < 25; ++kk) { v[kk] = __expf(v[kk] - mx); s += v[kk]; }
    const float inv = 1.0f / s;

    float* dst = Wn + (((size_t)b * 256 + (2 * y + si)) * 256 + (2 * x + sj)) * 25;
    #pragma unroll
    for (int kk = 0; kk < 25; ++kk) dst[kk] = v[kk] * inv;
}

// ---------------------------------------------------------------------------
// Kernel 4: bilinear x2 upsample (half-pixel centers -> 0.25/0.75 blends,
// edge-clamped) fused with the 25-tap content-aware reassembly.
// Thread = one output pixel x 32 channels. X (64 MB) is L2-resident.
// ---------------------------------------------------------------------------
__global__ __launch_bounds__(256) void carafe_reassemble(
    const float* __restrict__ X, const float* __restrict__ Wn,
    float* __restrict__ out)
{
    const int tid = blockIdx.x * blockDim.x + threadIdx.x;   // 2^21 threads
    const int x_ = tid & 255;
    const int y_ = (tid >> 8) & 255;
    const int cg = (tid >> 16) & 7;
    const int b  = tid >> 19;

    const int py = y_ & 1, yb = y_ >> 1;
    const int px = x_ & 1, xb = x_ >> 1;
    int R[6], C[6];
    #pragma unroll
    for (int d = 0; d < 6; ++d) {
        int r = yb - 3 + py + d; R[d] = r < 0 ? 0 : (r > 127 ? 127 : r);
        int c = xb - 3 + px + d; C[d] = c < 0 ? 0 : (c > 127 ? 127 : c);
    }
    const float wy0 = py ? 0.75f : 0.25f, wy1 = 1.0f - wy0;
    const float wx0 = px ? 0.75f : 0.25f, wx1 = 1.0f - wx0;

    float wt[25];
    const float* wp = Wn + (((size_t)b * 256 + y_) * 256 + x_) * 25;
    #pragma unroll
    for (int k = 0; k < 25; ++k) wt[k] = wp[k];

    const int c0 = cg << 5;
    const size_t opix = (size_t)y_ * 256 + x_;
    for (int c = c0; c < c0 + 32; ++c) {
        const float* xc = X + ((size_t)b * 256 + c) * 16384;
        float t[6][6];
        #pragma unroll
        for (int r = 0; r < 6; ++r)
            #pragma unroll
            for (int q = 0; q < 6; ++q)
                t[r][q] = xc[R[r] * 128 + C[q]];
        float cx[6][5];
        #pragma unroll
        for (int r = 0; r < 6; ++r)
            #pragma unroll
            for (int e = 0; e < 5; ++e)
                cx[r][e] = wx0 * t[r][e] + wx1 * t[r][e + 1];
        float acc = 0.0f;
        #pragma unroll
        for (int d = 0; d < 5; ++d)
            #pragma unroll
            for (int e = 0; e < 5; ++e) {
                const float u = wy0 * cx[d][e] + wy1 * cx[d + 1][e];
                acc = fmaf(wt[d * 5 + e], u, acc);
            }
        out[((size_t)b * 256 + c) * 65536 + opix] = acc;
    }
}

// ---------------------------------------------------------------------------
extern "C" void kernel_launch(void* const* d_in, const int* in_sizes, int n_in,
                              void* d_out, int out_size, void* d_ws, size_t ws_size,
                              hipStream_t stream)
{
    const float* X      = (const float*)d_in[0];   // (4,256,128,128)
    const float* comp_w = (const float*)d_in[1];   // (64,256)
    const float* comp_b = (const float*)d_in[2];   // (64)
    const float* enc_w  = (const float*)d_in[3];   // (100,576)
    const float* enc_b  = (const float*)d_in[4];   // (100)
    float* out = (float*)d_out;                    // (4,256,256,256)

    float* Wc = (float*)d_ws;                        // 4*64*16384   f32 (16 MB)
    float* We = Wc + (size_t)4 * 64 * 16384;         // 4*100*16384  f32 (26 MB)
    float* Wn = We + (size_t)4 * 100 * 16384;        // 4*65536*25   f32 (26 MB)

    // 1x1 conv: 4 batch * 4 Mtiles * 1024 Ntiles = 16384 waves, 8 waves/block
    carafe_conv1x1_wmma<<<2048, 256, 0, stream>>>(X, comp_w, comp_b, Wc);
    // 3x3 conv: 4 * 7 * 1024 = 28672 waves
    carafe_conv3x3_wmma<<<3584, 256, 0, stream>>>(Wc, enc_w, enc_b, We);
    // softmax: 262144 threads
    carafe_softmax<<<1024, 256, 0, stream>>>(We, Wn);
    // reassembly: 2097152 threads
    carafe_reassemble<<<8192, 256, 0, stream>>>(X, Wn, out);
}
